// InnerProductDecoder_3040836846104
// MI455X (gfx1250) — matrix operand order repaired and checked by reading, not compiled, not run on hardware
//
#include <hip/hip_runtime.h>

// ---------------------------------------------------------------------------
// Types for WMMA fragments (wave32, CDNA5)
// ---------------------------------------------------------------------------
typedef _Float16 v16h __attribute__((ext_vector_type(16)));
typedef _Float16 v8h  __attribute__((ext_vector_type(8)));
typedef _Float16 v4h  __attribute__((ext_vector_type(4)));
typedef float    v8f  __attribute__((ext_vector_type(8)));
typedef float    v4f  __attribute__((ext_vector_type(4)));

typedef __attribute__((address_space(3))) _Float16 lds_h;   // 32-bit LDS pointer

#define D_MODEL 512
#define NGRAPH  64
#define MAXN    512
#define KCH     32     // K-chunk per LDS stage (one WMMA K-step)
#define LDT     40     // LDS row stride in halves (32 + 8 pad, 80 B, 16B-aligned)

// ---------------------------------------------------------------------------
// Kernel 1: zero-fill the padded f16 buffer (vectorized, 16 B per thread)
// ---------------------------------------------------------------------------
__global__ __launch_bounds__(256) void zero_pad_kernel(_Float16* __restrict__ pad) {
  size_t i = (size_t)blockIdx.x * blockDim.x + threadIdx.x;
  v8h z = {};
  ((v8h*)pad)[i] = z;
}

// ---------------------------------------------------------------------------
// Kernel 2: scatter nodes into padded[g][p][:], converting f32 -> f16
// ---------------------------------------------------------------------------
__global__ __launch_bounds__(128) void scatter_kernel(const float* __restrict__ h,
                                                      const int*   __restrict__ gid,
                                                      const int*   __restrict__ pid,
                                                      _Float16*    __restrict__ pad) {
  const int node = blockIdx.x;
  const int c    = threadIdx.x * 4;
  const int g    = gid[node];
  const int p    = pid[node];
  v4f s = *(const v4f*)&h[(size_t)node * D_MODEL + c];
  v4h d = { (_Float16)s.x, (_Float16)s.y, (_Float16)s.z, (_Float16)s.w };
  *(v4h*)&pad[((size_t)g * MAXN + p) * D_MODEL + c] = d;
}

// ---------------------------------------------------------------------------
// Async copy of one 128x32 f16 chunk pair into LDS (ASYNCcnt path, GVS mode:
// scalar base advances per chunk, per-lane 32-bit offsets are loop-invariant).
// 256 threads x 2 iters x 2 tiles, 16 B per lane per instruction.
// ---------------------------------------------------------------------------
__device__ __forceinline__ void issue_chunk(const _Float16* sbase,      // Pb + kb
                                            const int voffA[2], const int voffB[2],
                                            lds_h* const ldsA[2], lds_h* const ldsB[2]) {
#pragma unroll
  for (int i = 0; i < 2; ++i) {
    asm volatile("global_load_async_to_lds_b128 %0, %1, %2"
                 :: "v"(ldsA[i]), "v"(voffA[i]), "s"(sbase) : "memory");
    asm volatile("global_load_async_to_lds_b128 %0, %1, %2"
                 :: "v"(ldsB[i]), "v"(voffB[i]), "s"(sbase) : "memory");
  }
}

__device__ __forceinline__ void async_wait_barrier() {
  asm volatile("s_wait_asynccnt 0x0" ::: "memory");
  __syncthreads();
}

// ---------------------------------------------------------------------------
// 8 WMMAs on one staged 128x32 chunk pair (per wave: 32x64 sub-tile).
// ---------------------------------------------------------------------------
__device__ __forceinline__ void compute_chunk(const _Float16* bA, const _Float16* bB,
                                              v8f acc[2][4],
                                              int wm, int wn, int l15, int hi) {
  // A fragments: 16x32 f16 layout — lane m holds row m;
  // K = {hi*8 .. hi*8+7} then {16+hi*8 .. 16+hi*8+7}
  v16h af[2];
#pragma unroll
  for (int rt = 0; rt < 2; ++rt) {
    const _Float16* base = &bA[(wm * 32 + rt * 16 + l15) * LDT + hi * 8];
    v8h s0 = *(const v8h*)base;
    v8h s1 = *(const v8h*)(base + 16);
    af[rt] = __builtin_shufflevector(s0, s1, 0,1,2,3,4,5,6,7,8,9,10,11,12,13,14,15);
  }
  // B fragments: B = A2^T — lane j holds row (col j); K = {hi*8 .. hi*8+7} then {16+hi*8 .. 16+hi*8+7}
  v16h bf[4];
#pragma unroll
  for (int ct = 0; ct < 4; ++ct) {
    const _Float16* base = &bB[(wn * 64 + ct * 16 + l15) * LDT + hi * 8];
    v8h s0 = *(const v8h*)base;
    v8h s1 = *(const v8h*)(base + 16);
    bf[ct] = __builtin_shufflevector(s0, s1, 0,1,2,3,4,5,6,7,8,9,10,11,12,13,14,15);
  }
#pragma unroll
  for (int rt = 0; rt < 2; ++rt)
#pragma unroll
    for (int ct = 0; ct < 4; ++ct)
      acc[rt][ct] = __builtin_amdgcn_wmma_f32_16x16x32_f16(
          false, af[rt], false, bf[ct], (short)0, acc[rt][ct], false, false);
}

// ---------------------------------------------------------------------------
// Kernel 3: batched C = P * P^T, 128x128 tile per block, async double-buffered.
// ---------------------------------------------------------------------------
__global__ __launch_bounds__(256) void syrk_wmma_kernel(const _Float16* __restrict__ P,
                                                        float* __restrict__ out) {
  __shared__ _Float16 sA[2][128 * LDT];   // 2 x 10 KB
  __shared__ _Float16 sB[2][128 * LDT];   // 2 x 10 KB

  const int blk = blockIdx.x;
  const int b   = blk >> 4;
  const int tr  = (blk >> 2) & 3;
  const int tc  = blk & 3;

  const _Float16* __restrict__ Pb = P + (size_t)b * (MAXN * D_MODEL);
  float* __restrict__           Ob = out + (size_t)b * (MAXN * MAXN);

  const int r0   = tr * 128;
  const int c0   = tc * 128;
  const int tid  = threadIdx.x;
  const int lane = tid & 31;
  const int wave = tid >> 5;
  const int wm   = wave & 3;        // 0..3 -> 32-row band
  const int wn   = wave >> 2;       // 0..1 -> 64-col band
  const int l15  = lane & 15;
  const int hi   = lane >> 4;

  // Loop-invariant per-thread async-copy operands -----------------------------
  int    voffA[2][2], voffB[2][2];            // [dummy buf dim folded] -> use [2]
  int    vA[2], vB[2];
  lds_h* lA[2][2];
  lds_h* lB[2][2];
#pragma unroll
  for (int i = 0; i < 2; ++i) {
    int v   = tid + i * 256;                  // 0..511
    int row = v >> 2;                         // 4 x 16B vectors per 64 B row
    int cv  = (v & 3) << 3;                   // column offset in halves
    vA[i] = (int)(((r0 + row) * D_MODEL + cv) * sizeof(_Float16));
    vB[i] = (int)(((c0 + row) * D_MODEL + cv) * sizeof(_Float16));
#pragma unroll
    for (int bi = 0; bi < 2; ++bi) {
      lA[bi][i] = (lds_h*)&sA[bi][row * LDT + cv];
      lB[bi][i] = (lds_h*)&sB[bi][row * LDT + cv];
    }
  }
  (void)voffA; (void)voffB;

  v8f acc[2][4] = {};

  // Prologue: prefetch chunk 0 into buffer 0
  issue_chunk(Pb, vA, vB, lA[0], lB[0]);

  // Software pipeline, 2 chunks (buf0, buf1) per loop iteration
  for (int kb = 0; kb < D_MODEL; kb += 2 * KCH) {
    async_wait_barrier();                                   // chunk kb ready (buf0)
    issue_chunk(Pb + kb + KCH, vA, vB, lA[1], lB[1]);       // prefetch -> buf1
    compute_chunk(sA[0], sB[0], acc, wm, wn, l15, hi);

    async_wait_barrier();                                   // chunk kb+32 ready (buf1)
    if (kb + 2 * KCH < D_MODEL)
      issue_chunk(Pb + kb + 2 * KCH, vA, vB, lA[0], lB[0]); // prefetch -> buf0
    compute_chunk(sA[1], sB[1], acc, wm, wn, l15, hi);
  }

  // Store: f32 16x16 C/D layout — VGPR r: lanes 0-15 -> M=r, lanes 16-31 -> M=r+8
#pragma unroll
  for (int rt = 0; rt < 2; ++rt) {
#pragma unroll
    for (int ct = 0; ct < 4; ++ct) {
      const int col   = c0 + wn * 64 + ct * 16 + l15;
      const int rbase = r0 + wm * 32 + rt * 16 + hi * 8;
#pragma unroll
      for (int r = 0; r < 8; ++r)
        Ob[(size_t)(rbase + r) * MAXN + col] = acc[rt][ct][r];
    }
  }
}

// ---------------------------------------------------------------------------
// Host-side launch
// ---------------------------------------------------------------------------
extern "C" void kernel_launch(void* const* d_in, const int* in_sizes, int n_in,
                              void* d_out, int out_size, void* d_ws, size_t ws_size,
                              hipStream_t stream) {
  const float* batched_h = (const float*)d_in[0];
  const int*   graph_ids = (const int*)d_in[1];
  const int*   pos_ids   = (const int*)d_in[2];
  const int    nodes     = in_sizes[1];          // element count of graph_ids

  _Float16* pad = (_Float16*)d_ws;               // 64*512*512*2 = 32 MB scratch
  float*    out = (float*)d_out;                 // 64*512*512 f32

  const int nvec = (NGRAPH * MAXN * D_MODEL) / 8;
  zero_pad_kernel<<<nvec / 256, 256, 0, stream>>>(pad);
  scatter_kernel<<<nodes, 128, 0, stream>>>(batched_h, graph_ids, pos_ids, pad);
  syrk_wmma_kernel<<<NGRAPH * 16, 256, 0, stream>>>(pad, out);
}
